// EncoderLayer_63934883168653
// MI455X (gfx1250) — matrix operand order repaired and checked by reading, not compiled, not run on hardware
//
#include <hip/hip_runtime.h>
#include <hip/hip_bf16.h>

// ---------------------------------------------------------------------------
// Encoder layer on MI455X (gfx1250): bf16 WMMA everywhere.
// B=8 L=1024 D=1024 H=16 DH=64 DFF=4096
// ---------------------------------------------------------------------------

typedef __attribute__((ext_vector_type(16))) __bf16 v16bf;
typedef __attribute__((ext_vector_type(8)))  __bf16 v8bf;
typedef __attribute__((ext_vector_type(8)))  float  v8f;
typedef int v4i_vec __attribute__((vector_size(16)));   // matches builtin proto

#define EB 8
#define EL 1024
#define ED 1024
#define EH 16
#define EDH 64
#define EDFF 4096
#define EM (EB * EL)   // 8192 rows
#define MW (EL / 32)   // mask words per row

// --- gfx1250 async global->LDS copy (guarded; falls back to ld+ds_store) ----
#define HAVE_ASYNC_LDS 0
#if defined(__has_builtin)
#if __has_builtin(__builtin_amdgcn_global_load_async_to_lds_b128)
#undef HAVE_ASYNC_LDS
#define HAVE_ASYNC_LDS 1
#endif
#endif

__device__ __forceinline__ void cp16_g2l(const __bf16* g, __bf16* l) {
#if HAVE_ASYNC_LDS
    __builtin_amdgcn_global_load_async_to_lds_b128(
        (v4i_vec*)g, (v4i_vec*)l, 0, 0);
#else
    *(uint4*)l = *(const uint4*)g;
#endif
}

template <int N>
__device__ __forceinline__ void async_wait() {
#if HAVE_ASYNC_LDS
#if defined(__has_builtin) && __has_builtin(__builtin_amdgcn_s_wait_asynccnt)
    __builtin_amdgcn_s_wait_asynccnt(N);
#else
    asm volatile("s_wait_asynccnt %0" :: "i"(N) : "memory");
#endif
#endif
}

// D = A(16x32 bf16) x B(32x16 bf16) + C(16x16 f32)
__device__ __forceinline__ v8f wmma_bf16(v16bf a, v16bf b, v8f c) {
    return __builtin_amdgcn_wmma_f32_16x16x32_bf16(
        /*neg_a=*/false, a, /*neg_b=*/false, b,
        /*c_mod=*/(short)0, c, /*reuse_a=*/false, /*reuse_b=*/false);
}

// A-fragment (16x32, row-major source): lane l holds row m=l%16;
// element j maps to k = (l/16)*16 + j  -> two 16B runs.
__device__ __forceinline__ v16bf load_a_frag(const __bf16* row, int kbase, int half) {
    v8bf r0 = *(const v8bf*)(row + kbase + half * 16);
    v8bf r1 = *(const v8bf*)(row + kbase + half * 16 + 8);
    return __builtin_shufflevector(r0, r1, 0,1,2,3,4,5,6,7,8,9,10,11,12,13,14,15);
}

// B-fragment (32x16): lane l holds column n=l%16, k = (l/16)*16 + j.
// Caller passes pointer to the 16-element contiguous run (16B aligned).
__device__ __forceinline__ v16bf load_b_frag(const __bf16* run) {
    return *(const v16bf*)run;
}

__device__ __forceinline__ float wave_sum(float v) {
    #pragma unroll
    for (int o = 16; o > 0; o >>= 1) v += __shfl_xor(v, o, 32);
    return v;
}

// ------------------------------ LayerNorm -> bf16 ---------------------------
__global__ void ln_bf16_kernel(const float* __restrict__ x,
                               const float* __restrict__ g,
                               const float* __restrict__ be,
                               __bf16* __restrict__ out) {
    const int row = blockIdx.x;            // 0..EM-1
    const float* xr = x + (size_t)row * ED;
    const int tid = threadIdx.x;           // 256 threads
    float v[4];
    float s = 0.f;
    #pragma unroll
    for (int i = 0; i < 4; ++i) { v[i] = xr[tid + 256 * i]; s += v[i]; }
    __shared__ float red[8];
    s = wave_sum(s);
    const int lane = tid & 31, wid = tid >> 5;
    if (lane == 0) red[wid] = s;
    __syncthreads();
    float tot = red[0] + red[1] + red[2] + red[3] + red[4] + red[5] + red[6] + red[7];
    const float mu = tot * (1.f / ED);
    float s2 = 0.f;
    #pragma unroll
    for (int i = 0; i < 4; ++i) { float d = v[i] - mu; s2 += d * d; }
    s2 = wave_sum(s2);
    __syncthreads();
    if (lane == 0) red[wid] = s2;
    __syncthreads();
    float tot2 = red[0] + red[1] + red[2] + red[3] + red[4] + red[5] + red[6] + red[7];
    const float rstd = rsqrtf(tot2 * (1.f / ED) + 1e-5f);
    __bf16* orow = out + (size_t)row * ED;
    #pragma unroll
    for (int i = 0; i < 4; ++i) {
        int col = tid + 256 * i;
        orow[col] = (__bf16)((v[i] - mu) * rstd * g[col] + be[col]);
    }
}

// ------------------ weight cast+transpose: W[K,N]f32 -> Wt[N,K]bf16 ---------
__global__ void wtrans_kernel(const float* __restrict__ W, __bf16* __restrict__ Wt,
                              int K, int N) {
    size_t idx = (size_t)blockIdx.x * blockDim.x + threadIdx.x;
    if (idx >= (size_t)K * N) return;
    size_t k = idx / N;
    size_t n = idx % N;
    Wt[n * (size_t)K + k] = (__bf16)W[idx];
}

// ---------------- q head transpose: q[B,L,D] -> qt[B*H, DH, L] --------------
__global__ void qtrans_kernel(const __bf16* __restrict__ q, __bf16* __restrict__ qt) {
    size_t idx = (size_t)blockIdx.x * blockDim.x + threadIdx.x;
    if (idx >= (size_t)EB * EL * ED) return;
    int d = (int)(idx & (ED - 1));
    size_t bl = idx >> 10;
    int l = (int)(bl & (EL - 1));
    int b = (int)(bl >> 10);
    int h = d >> 6, dh = d & 63;
    qt[(((size_t)(b * EH + h) * EDH) + dh) * EL + l] = q[idx];
}

// --------- mask pack: bool[B,L,L] -> u32 bitmask [B, L, L/32] ---------------
__global__ void maskbits_kernel(const unsigned char* __restrict__ mask,
                                unsigned int* __restrict__ bits) {
    size_t idx = (size_t)blockIdx.x * blockDim.x + threadIdx.x;
    if (idx >= (size_t)EB * EL * MW) return;
    union { uint4 v; unsigned char c[16]; } u0, u1;
    const uint4* p4 = (const uint4*)(mask + idx * 32);
    u0.v = p4[0];
    u1.v = p4[1];
    unsigned int m = 0;
    #pragma unroll
    for (int i = 0; i < 16; ++i) {
        m |= (u0.c[i] ? 1u : 0u) << i;
        m |= (u1.c[i] ? 1u : 0u) << (16 + i);
    }
    bits[idx] = m;
}

// ---------------------------- generic WMMA GEMM -----------------------------
// Wave tile 32(m) x 64(n): 2 A-frags x 4 B-frags -> 8 WMMA per k-step of 32.
// Block: 8 waves = 4(m) x 2(n) -> 128 x 128 block tile.
// KDIM/MODE are compile-time: all operand addresses fold into one A base and
// one B base register + 24-bit immediates (no per-iteration 64-bit VALU).
// MODE 0: outb = val*scale; 1: outf = val+resid; 2: outb = relu(val+bias);
// MODE 3: outf = val+bias+resid
template <int KDIM, int MODE>
__global__ void gemm_bf16_kernel(const __bf16* __restrict__ A,
                                 const __bf16* __restrict__ Bt,
                                 float* __restrict__ outf,
                                 __bf16* __restrict__ outb,
                                 const float* __restrict__ bias,
                                 const float* __restrict__ resid,
                                 int Ndim, float scale) {
    const int lane = threadIdx.x & 31;
    const int w    = threadIdx.x >> 5;     // 8 waves: 4 (m) x 2 (n)
    const int wm = w & 3, wn = w >> 2;
    const int half = lane >> 4, ln16 = lane & 15;
    const int m0 = blockIdx.x * 128 + wm * 32;
    const int n0 = blockIdx.y * 128 + wn * 64;

    v8f acc[2][4];
    #pragma unroll
    for (int i = 0; i < 2; ++i)
        #pragma unroll
        for (int j = 0; j < 4; ++j)
            #pragma unroll
            for (int r = 0; r < 8; ++r) acc[i][j][r] = 0.f;

    const __bf16* abase = A + (size_t)(m0 + ln16) * KDIM;
    const __bf16* bbase = Bt + (size_t)(n0 + ln16) * KDIM + half * 16;

    for (int k = 0; k < KDIM; k += 32) {
        __builtin_prefetch(bbase + k + 256, 0, 0);
        __builtin_prefetch(bbase + 2 * 16 * KDIM + k + 256, 0, 0);
        v16bf a0 = load_a_frag(abase, k, half);
        v16bf a1 = load_a_frag(abase + 16 * KDIM, k, half);
        #pragma unroll
        for (int j = 0; j < 4; ++j) {
            v16bf b = load_b_frag(bbase + j * 16 * KDIM + k);
            acc[0][j] = wmma_bf16(a0, b, acc[0][j]);
            acc[1][j] = wmma_bf16(a1, b, acc[1][j]);
        }
    }

    #pragma unroll
    for (int i = 0; i < 2; ++i) {
        #pragma unroll
        for (int j = 0; j < 4; ++j) {
            const int mb = m0 + i * 16 + half * 8;
            const int nb = n0 + j * 16 + ln16;
            #pragma unroll
            for (int r = 0; r < 8; ++r) {
                const size_t idx = (size_t)(mb + r) * Ndim + nb;
                const float v = acc[i][j][r];
                if (MODE == 0)      outb[idx] = (__bf16)(v * scale);
                else if (MODE == 1) outf[idx] = v + resid[idx];
                else if (MODE == 2) outb[idx] = (__bf16)fmaxf(v + bias[nb], 0.f);
                else                outf[idx] = v + bias[nb] + resid[idx];
            }
        }
    }
}

// ------------------------- flash attention (per head) -----------------------
// 4 waves/block share one (b,h); K/V key-blocks double-buffered in LDS via
// async global->LDS (issue next block, s_wait_asynccnt 4 overlaps DMA with
// WMMA compute). Each wave owns 16 query rows. Mask comes as row bitmasks.
#define KPAD 72   // 32 x 64 K-tile, padded row stride (elements)
#define VPAD 40   // 64 x 32 V-tile, padded row stride (elements)
__global__ void attn_kernel(const __bf16* __restrict__ Qs,
                            const __bf16* __restrict__ Kbf,
                            const __bf16* __restrict__ Vt,
                            const unsigned int* __restrict__ mbits,
                            __bf16* __restrict__ ctx) {
    __shared__ alignas(64) __bf16 kblk[2][32 * KPAD];  // [key][dh]
    __shared__ alignas(64) __bf16 vblk[2][64 * VPAD];  // [dh][key]
    __shared__ alignas(64) __bf16 plds[4][16 * 32];    // per-wave P tile

    const int tid  = threadIdx.x;        // 128
    const int lane = tid & 31;
    const int w    = tid >> 5;           // 4 waves
    const int half = lane >> 4, ln16 = lane & 15;
    const int bh = blockIdx.x;           // 0..B*H-1
    const int b = bh >> 4, h = bh & 15;
    const int q0 = blockIdx.y * 64 + w * 16;

    // pq A-fragments (16x64 -> two 16x32 fragments), loaded once
    const __bf16* arow = Qs + ((size_t)(b * EL + q0 + ln16) * ED + h * EDH);
    const v16bf a0 = load_a_frag(arow, 0, half);
    const v16bf a1 = load_a_frag(arow, 32, half);

    float mrow[8], lrow[8];
    v8f acc[4];
    #pragma unroll
    for (int r = 0; r < 8; ++r) { mrow[r] = -1e30f; lrow[r] = 0.f; }
    #pragma unroll
    for (int c = 0; c < 4; ++c)
        #pragma unroll
        for (int r = 0; r < 8; ++r) acc[c][r] = 0.f;

    const __bf16* kbase = Kbf + (size_t)b * EL * ED + h * EDH;   // + key*ED
    const __bf16* vbase = Vt + (size_t)bh * EDH * EL;            // + dh*EL
    // bitmask rows this lane owns: q0 + half*8 + r
    const unsigned int* mrow_bits =
        mbits + ((size_t)b * EL + q0 + half * 8) * MW;
    __bf16* pl = plds[w];

    // stage one 32-key block (4 async instructions per wave)
    auto stage = [&](int kbs, __bf16* kd, __bf16* vd) {
        #pragma unroll
        for (int it = 0; it < 2; ++it) {
            const int ch = tid + it * 128;          // 0..255
            const int kr = ch >> 3, kc = ch & 7;    // K: 32 rows x 8 chunks
            cp16_g2l(kbase + (size_t)(kbs + kr) * ED + kc * 8,
                     kd + kr * KPAD + kc * 8);
            const int vr = ch >> 2, vc = ch & 3;    // V: 64 rows x 4 chunks
            cp16_g2l(vbase + (size_t)vr * EL + kbs + vc * 8,
                     vd + vr * VPAD + vc * 8);
        }
    };

    stage(0, kblk[0], vblk[0]);
    int cur = 0;

    for (int kb = 0; kb < EL; kb += 32) {
        // ---- prefetch next block, then wait for current (in-order async) ----
        if (kb + 32 < EL) {
            stage(kb + 32, kblk[cur ^ 1], vblk[cur ^ 1]);
            async_wait<4>();
        } else {
            async_wait<0>();
        }
        __syncthreads();
        const __bf16* kB = kblk[cur];
        const __bf16* vB = vblk[cur];

        // ---- issue mask-word loads early (latency hidden under WMMAs) ----
        unsigned int mw[8];
        #pragma unroll
        for (int r = 0; r < 8; ++r) mw[r] = mrow_bits[r * MW + (kb >> 5)];

        // ---- S = pq @ pk^T for 32 keys (two 16x16 tiles) ----
        v8f s0, s1;
        #pragma unroll
        for (int r = 0; r < 8; ++r) { s0[r] = 0.f; s1[r] = 0.f; }
        {
            const __bf16* k0 = kB + ln16 * KPAD + half * 16;
            const __bf16* k1 = kB + (16 + ln16) * KPAD + half * 16;
            s0 = wmma_bf16(a0, load_b_frag(k0),      s0);
            s0 = wmma_bf16(a1, load_b_frag(k0 + 32), s0);
            s1 = wmma_bf16(a0, load_b_frag(k1),      s1);
            s1 = wmma_bf16(a1, load_b_frag(k1 + 32), s1);
        }
        // ---- mask via bit tests (no serialized byte loads) ----
        #pragma unroll
        for (int r = 0; r < 8; ++r) {
            s0[r] = ((mw[r] >> ln16) & 1u)        ? -1e30f : s0[r];
            s1[r] = ((mw[r] >> (16 + ln16)) & 1u) ? -1e30f : s1[r];
        }
        // ---- online softmax (row reductions across 16-lane halves) ----
        float ps0[8], ps1[8];
        #pragma unroll
        for (int r = 0; r < 8; ++r) {
            float cur_m = fmaxf(s0[r], s1[r]);
            #pragma unroll
            for (int o = 1; o < 16; o <<= 1) cur_m = fmaxf(cur_m, __shfl_xor(cur_m, o, 32));
            const float mnew = fmaxf(mrow[r], cur_m);
            const float alpha = __expf(mrow[r] - mnew);
            mrow[r] = mnew;
            const float p0 = __expf(s0[r] - mnew);
            const float p1 = __expf(s1[r] - mnew);
            float rs = p0 + p1;
            #pragma unroll
            for (int o = 1; o < 16; o <<= 1) rs += __shfl_xor(rs, o, 32);
            lrow[r] = lrow[r] * alpha + rs;
            acc[0][r] *= alpha; acc[1][r] *= alpha;
            acc[2][r] *= alpha; acc[3][r] *= alpha;
            ps0[r] = p0; ps1[r] = p1;
        }
        // ---- P (C-layout f32) -> LDS -> A-layout bf16 fragment ----
        #pragma unroll
        for (int r = 0; r < 8; ++r) {
            pl[(half * 8 + r) * 32 + ln16]      = (__bf16)ps0[r];
            pl[(half * 8 + r) * 32 + 16 + ln16] = (__bf16)ps1[r];
        }
        __syncthreads();
        const v16bf pa = load_a_frag(pl + ln16 * 32, 0, half);
        // ---- O += P @ pv (DH=64 -> 4 n-chunks), V from LDS ----
        #pragma unroll
        for (int c = 0; c < 4; ++c) {
            const __bf16* vp = vB + (c * 16 + ln16) * VPAD + half * 16;
            acc[c] = wmma_bf16(pa, load_b_frag(vp), acc[c]);
        }
        __syncthreads();   // all reads done before buffer is restaged
        cur ^= 1;
    }

    // ---- finalize: O /= l ----
    #pragma unroll
    for (int r = 0; r < 8; ++r) {
        const float inv = 1.f / lrow[r];
        const int row = q0 + half * 8 + r;
        __bf16* op = ctx + ((size_t)(b * EL + row) * ED + h * EDH);
        #pragma unroll
        for (int c = 0; c < 4; ++c)
            op[c * 16 + ln16] = (__bf16)(acc[c][r] * inv);
    }
}

// ---------------------------------------------------------------------------
extern "C" void kernel_launch(void* const* d_in, const int* in_sizes, int n_in,
                              void* d_out, int out_size, void* d_ws, size_t ws_size,
                              hipStream_t stream) {
    (void)in_sizes; (void)n_in; (void)out_size; (void)ws_size;

    const float* x     = (const float*)d_in[0];
    const unsigned char* mask = (const unsigned char*)d_in[1];   // bool
    const float* Wq    = (const float*)d_in[2];
    const float* Wo    = (const float*)d_in[3];
    const float* W1    = (const float*)d_in[4];
    const float* b1    = (const float*)d_in[5];
    const float* W2    = (const float*)d_in[6];
    const float* b2    = (const float*)d_in[7];
    const float* ln1_g = (const float*)d_in[8];
    const float* ln1_b = (const float*)d_in[9];
    const float* ln2_g = (const float*)d_in[10];
    const float* ln2_b = (const float*)d_in[11];
    float* out = (float*)d_out;

    char* ws = (char*)d_ws;
    const size_t MB = 1024ull * 1024ull;
    __bf16* q_bf = (__bf16*)(ws);                 // 16 MB  [B,L,D]
    __bf16* qt   = (__bf16*)(ws + 16 * MB);       // 16 MB  [B*H,DH,L]
    __bf16* Qs   = (__bf16*)(ws + 32 * MB);       // 16 MB  scaled pq [B,L,D]
    __bf16* ctx  = (__bf16*)(ws + 48 * MB);       // 16 MB  [B,L,D]
    float*  x1   = (float*) (ws + 64 * MB);       // 32 MB  residual after attn
    __bf16* h_bf = (__bf16*)(ws + 96 * MB);       // 16 MB  LN2 output
    __bf16* act  = (__bf16*)(ws + 112 * MB);      // 64 MB  FFN hidden [M,DFF]
    __bf16* Wq_t = (__bf16*)(ws + 176 * MB);      //  2 MB
    __bf16* Wo_t = (__bf16*)(ws + 178 * MB);      //  2 MB
    __bf16* W1_t = (__bf16*)(ws + 180 * MB);      //  8 MB
    __bf16* W2_t = (__bf16*)(ws + 188 * MB);      //  8 MB
    unsigned int* mbits = (unsigned int*)(ws + 196 * MB);  // 1 MB

    // 1. q = LN1(x), cast bf16
    ln_bf16_kernel<<<EM, 256, 0, stream>>>(x, ln1_g, ln1_b, q_bf);

    // 2. layout transforms (weights once, q head-transpose, mask bitpack)
    {
        size_t n;
        n = (size_t)ED * ED;
        wtrans_kernel<<<(int)((n + 255) / 256), 256, 0, stream>>>(Wq, Wq_t, ED, ED);
        wtrans_kernel<<<(int)((n + 255) / 256), 256, 0, stream>>>(Wo, Wo_t, ED, ED);
        n = (size_t)ED * EDFF;
        wtrans_kernel<<<(int)((n + 255) / 256), 256, 0, stream>>>(W1, W1_t, ED, EDFF);
        wtrans_kernel<<<(int)((n + 255) / 256), 256, 0, stream>>>(W2, W2_t, EDFF, ED);
        n = (size_t)EB * EL * ED;
        qtrans_kernel<<<(int)((n + 255) / 256), 256, 0, stream>>>(q_bf, qt);
        n = (size_t)EB * EL * MW;
        maskbits_kernel<<<(int)((n + 255) / 256), 256, 0, stream>>>(mask, mbits);
    }

    // 3. Qs = (q @ Wq) * DH^-0.5      [M,D]
    {
        dim3 grid(EM / 128, ED / 128);
        gemm_bf16_kernel<ED, 0><<<grid, 256, 0, stream>>>(
            q_bf, Wq_t, nullptr, Qs, nullptr, nullptr, ED, 0.125f);
    }

    // 4. flash attention: ctx[B,L,D]
    {
        dim3 grid(EB * EH, EL / 64);
        attn_kernel<<<grid, 128, 0, stream>>>(Qs, q_bf, qt, mbits, ctx);
    }

    // 5. x1 = ctx @ Wo + x
    {
        dim3 grid(EM / 128, ED / 128);
        gemm_bf16_kernel<ED, 1><<<grid, 256, 0, stream>>>(
            ctx, Wo_t, x1, nullptr, nullptr, x, ED, 1.0f);
    }

    // 6. h = LN2(x1), cast bf16
    ln_bf16_kernel<<<EM, 256, 0, stream>>>(x1, ln2_g, ln2_b, h_bf);

    // 7. act = relu(h @ W1 + b1)
    {
        dim3 grid(EM / 128, EDFF / 128);
        gemm_bf16_kernel<ED, 2><<<grid, 256, 0, stream>>>(
            h_bf, W1_t, nullptr, act, b1, nullptr, EDFF, 1.0f);
    }

    // 8. out = act @ W2 + b2 + x1
    {
        dim3 grid(EM / 128, ED / 128);
        gemm_bf16_kernel<EDFF, 3><<<grid, 256, 0, stream>>>(
            act, W2_t, out, nullptr, b2, x1, ED, 1.0f);
    }
}
